// GraphFP_16587163697388
// MI455X (gfx1250) — matrix operand order repaired and checked by reading, not compiled, not run on hardware
//
#include <hip/hip_runtime.h>

// CDNA5 / gfx1250 fused GraphFP kernel.
// One workgroup (256 threads = 8 wave32) per batch element.
// Stage 1: stream graph slice (541KB) -> neigh_sum + diagonal (LDS)
// Stage 2: per branch: core (VALU f32, exact) -> logits via
//          v_wmma_f32_16x16x32_f16 with f16 hi/lo split (near-fp32 accuracy)
// Stage 3: softmax over OUT=512 + row-sum accumulation, cross-wave reduce.

typedef __attribute__((ext_vector_type(16))) _Float16 v16h;
typedef __attribute__((ext_vector_type(8)))  float    v8f;

#define B_   512
#define N_   64
#define F_   32
#define OUT_ 512
#define GCH  33   // F+1 channels in graph

// LDS layout in floats
#define L_LOGITS 0           // 64*512  = 32768 floats
#define L_NEIGH  32768       // 64*32   = 2048
#define L_ATTR   34816       // 64*32   = 2048
#define L_CORE   36864       // 64*32   = 2048
#define SMEM_FLOATS 38912    // 155,648 bytes  (< 320KB WGP LDS)

__launch_bounds__(256, 1)
__global__ void graphfp_fused(const float* __restrict__ graph,
                              const float* __restrict__ W_inner,
                              const float* __restrict__ b_inner,
                              const float* __restrict__ W_output,
                              const float* __restrict__ b_output,
                              float* __restrict__ out)
{
    __shared__ float smem[SMEM_FLOATS];
    const int tid  = threadIdx.x;
    const int lane = tid & 31;
    const int wave = tid >> 5;
    const int b    = blockIdx.x;
    const int grp  = lane >> 4;     // which 16-lane half
    const int l15  = lane & 15;

    float* s_log = smem + L_LOGITS;
    float* s_ng  = smem + L_NEIGH;
    float* s_at  = smem + L_ATTR;
    float* s_co  = smem + L_CORE;

    // ---------- Stage 1: neighbor-sum + diagonal (memory-bound pass) ----------
    // lane == feature index f (0..31); drop channel 32 (bond flag).
    {
        const float* gbase = graph + (size_t)b * (N_ * N_ * GCH);
        for (int i = wave * 8; i < wave * 8 + 8; ++i) {
            const float* rowp = gbase + (size_t)i * (N_ * GCH);
            float s = 0.f, a = 0.f;
            #pragma unroll 8
            for (int j = 0; j < N_; ++j) {
                float v = rowp[j * GCH + lane];   // coalesced 128B per j
                s += v;
                if (j == i) a = v;                // diagonal attribute
            }
            s_ng[i * F_ + lane] = s;
            s_at[i * F_ + lane] = a;
        }
    }
    __syncthreads();

    float acc[16];                                // fp partial: col = i*32+lane
    #pragma unroll
    for (int i = 0; i < 16; ++i) acc[i] = 0.f;

    for (int br = 0; br < 3; ++br) {
        // ---------- core matrix for this branch ----------
        const float* coreSrc;
        if (br == 0) {
            coreSrc = s_at;                       // raw diagonal attributes
        } else {
            // core = neigh_sum @ W_inner[br] + N * b_inner[br]  (exact f32)
            const float* Wi = W_inner + br * F_ * F_;
            const float* bi = b_inner + br * F_;
            const int i  = tid >> 2;
            const int f0 = (tid & 3) * 8;
            #pragma unroll
            for (int ff = 0; ff < 8; ++ff) {
                const int f = f0 + ff;
                float c = 64.f * bi[f];
                #pragma unroll 8
                for (int kk = 0; kk < F_; ++kk)
                    c += s_ng[i * F_ + kk] * Wi[kk * F_ + f];
                s_co[i * F_ + f] = c;
            }
            __syncthreads();
            coreSrc = s_co;
        }
        const int selOut = (br == 2) ? 1 : 0;     // faithful W_output reuse of slice 0
        const float* Wo = W_output + selOut * (F_ * OUT_);
        const float* bo = b_output + selOut * OUT_;

        // ---------- A fragments (16x32 f16, hi/lo split) for 4 M-tiles ----------
        v16h ah[4], al[4];
        #pragma unroll
        for (int mt = 0; mt < 4; ++mt) {
            const int m = mt * 16 + l15;          // A row M = lane&15
            #pragma unroll
            for (int j = 0; j < 8; ++j) {
                float a0 = coreSrc[m * F_ + grp * 8 + j];        // K = grp*8+j
                float a1 = coreSrc[m * F_ + 16 + grp * 8 + j];   // K = 16+grp*8+j
                _Float16 h0 = (_Float16)a0;
                _Float16 h1 = (_Float16)a1;
                ah[mt][j]     = h0;  al[mt][j]     = (_Float16)(a0 - (float)h0);
                ah[mt][8 + j] = h1;  al[mt][8 + j] = (_Float16)(a1 - (float)h1);
            }
        }

        // ---------- logits = core @ W_output + b_output via WMMA ----------
        #pragma unroll
        for (int q = 0; q < 4; ++q) {
            const int nt = wave * 4 + q;          // this wave's N-tiles
            const int n  = nt * 16 + l15;         // B col N = lane&15
            v16h bh, bl;
            #pragma unroll
            for (int j = 0; j < 16; ++j) {        // K = (j>>3)*16 + grp*8 + (j&7)
                float v = Wo[((j >> 3) * 16 + grp * 8 + (j & 7)) * OUT_ + n];
                _Float16 h = (_Float16)v;
                bh[j] = h;
                bl[j] = (_Float16)(v - (float)h);
            }
            const float biasn = bo[n];
            #pragma unroll
            for (int mt = 0; mt < 4; ++mt) {
                v8f c;
                #pragma unroll
                for (int vv = 0; vv < 8; ++vv) c[vv] = biasn;
                // split-precision accumulation: AhBh + AlBh + AhBl (f32 acc)
                c = __builtin_amdgcn_wmma_f32_16x16x32_f16(false, ah[mt], false, bh,
                                                           (short)0, c, false, false);
                c = __builtin_amdgcn_wmma_f32_16x16x32_f16(false, al[mt], false, bh,
                                                           (short)0, c, false, false);
                c = __builtin_amdgcn_wmma_f32_16x16x32_f16(false, ah[mt], false, bl,
                                                           (short)0, c, false, false);
                #pragma unroll
                for (int vv = 0; vv < 8; ++vv)    // D row = vv + 8*grp
                    s_log[(mt * 16 + grp * 8 + vv) * OUT_ + n] = c[vv];
            }
        }
        __syncthreads();

        // ---------- softmax over OUT + sum over rows (this wave's 8 rows) ------
        for (int r = wave * 8; r < wave * 8 + 8; ++r) {
            float vals[16];
            float mx = -3.4e38f;
            #pragma unroll
            for (int i = 0; i < 16; ++i) {
                vals[i] = s_log[r * OUT_ + i * 32 + lane];
                mx = fmaxf(mx, vals[i]);
            }
            #pragma unroll
            for (int o = 16; o > 0; o >>= 1)
                mx = fmaxf(mx, __shfl_xor(mx, o, 32));
            float se = 0.f;
            #pragma unroll
            for (int i = 0; i < 16; ++i) {
                vals[i] = __expf(vals[i] - mx);
                se += vals[i];
            }
            #pragma unroll
            for (int o = 16; o > 0; o >>= 1)
                se += __shfl_xor(se, o, 32);
            const float rs = 1.f / se;
            #pragma unroll
            for (int i = 0; i < 16; ++i)
                acc[i] += vals[i] * rs;
        }
        __syncthreads();                          // protect s_co / s_log reuse
    }

    // ---------- cross-wave reduction of fp partials (reuse logits LDS) --------
    float* s_fp = smem;                           // aliases s_log (done with it)
    #pragma unroll
    for (int i = 0; i < 16; ++i)
        s_fp[wave * OUT_ + i * 32 + lane] = acc[i];
    __syncthreads();
    #pragma unroll
    for (int cc = 0; cc < 2; ++cc) {
        const int c = tid + cc * 256;
        float s = 0.f;
        #pragma unroll
        for (int wv = 0; wv < 8; ++wv) s += s_fp[wv * OUT_ + c];
        out[(size_t)b * OUT_ + c] = s;
    }
}

extern "C" void kernel_launch(void* const* d_in, const int* in_sizes, int n_in,
                              void* d_out, int out_size, void* d_ws, size_t ws_size,
                              hipStream_t stream) {
    (void)in_sizes; (void)n_in; (void)d_ws; (void)ws_size; (void)out_size;
    const float* graph    = (const float*)d_in[0];
    const float* W_inner  = (const float*)d_in[1];
    const float* b_inner  = (const float*)d_in[2];
    const float* W_output = (const float*)d_in[3];
    const float* b_output = (const float*)d_in[4];
    float* out = (float*)d_out;

    graphfp_fused<<<B_, 256, 0, stream>>>(graph, W_inner, b_inner,
                                          W_output, b_output, out);
}
